// SPGNet_17918603559387
// MI455X (gfx1250) — matrix operand order repaired and checked by reading, not compiled, not run on hardware
//
#include <hip/hip_runtime.h>
#include <hip/hip_bf16.h>

// ---------------------------------------------------------------------------
// SPGNet for gfx1250: 3 chained kernels, one 16-row tile per wave32,
// bf16 WMMA (v_wmma_f32_16x16x32_bf16) for every level's pointwise GEMM.
// ---------------------------------------------------------------------------

typedef __bf16 v16bf __attribute__((ext_vector_type(16)));
typedef float  v8f   __attribute__((ext_vector_type(8)));

union BFrag { v16bf v; unsigned u[8]; };

__device__ __forceinline__ unsigned short f2bf(float x) {
  unsigned u = __builtin_bit_cast(unsigned, x);
  u += 0x7FFFu + ((u >> 16) & 1u);          // round-to-nearest-even
  return (unsigned short)(u >> 16);
}

// A-matrix fragment (16x32 bf16) per ISA layout:
// lane L: m = L&15, kh = L>>4; VGPR v<4 -> K = 8*kh + 2v ; v>=4 -> K = 16 + 8*kh + 2(v-4)
__device__ __forceinline__ v16bf load_a_frag(const unsigned short* Ap, int KP,
                                             int m, int kh, int kbase) {
  BFrag f;
  const unsigned short* row = Ap + m * KP + kbase + 8 * kh;
#pragma unroll
  for (int v = 0; v < 8; ++v) {
    int K = (v < 4) ? (2 * v) : (16 + 2 * (v - 4));
    f.u[v] = *(const unsigned*)(row + K);
  }
  return f.v;
}

// B-matrix fragment (32x16 bf16). Bp is packed pairs: u32 [kpair][16 cols],
// word (kp, n) = { W[2kp][n], W[2kp+1][n] }. lane L: n = L&15, kh = L>>4
// VGPR v<4 holds K = 8*kh + {2v, 2v+1}; v>=4 holds K = 16 + 8*kh + {2(v-4), 2(v-4)+1} at column n.
__device__ __forceinline__ v16bf load_b_frag(const unsigned* Bp, int kh, int n,
                                             int kpbase) {
  BFrag f;
  const unsigned* p = Bp + (kpbase + 4 * kh) * 16 + n;
#pragma unroll
  for (int v = 0; v < 8; ++v) f.u[v] = p[(v < 4 ? v : v + 4) * 16];
  return f.v;
}

#define WMMA_BF16(a, b, c) \
  __builtin_amdgcn_wmma_f32_16x16x32_bf16(false, (a), false, (b), (short)0, (c), false, false)

// child-offset tables from UP_OFFSETS, encoded as bitmasks (bit j = offset of child j)
#define DXM 0xD2
#define DYM 0xB4
#define DZM 0xE8

// ---------------------------------------------------------------------------
// Coarse level: N0=32768 rows. gather 64ch -> LN -> [16x64]@[64x48] -> logit(48->1)
// ---------------------------------------------------------------------------
struct __align__(16) ScratchC {
  union { float F[16][65]; float D[16][48]; };
  unsigned short A[16][64];
};

__global__ __launch_bounds__(256) void spg_coarse(
    const float* __restrict__ gc, const int* __restrict__ vi,
    const float* __restrict__ lnw, const float* __restrict__ lnb,
    const float* __restrict__ Wn, const float* __restrict__ bn,
    const float* __restrict__ Wo, const float* __restrict__ bo,
    float* __restrict__ vfc, float* __restrict__ out_logc) {
  __shared__ unsigned Bp[3 * 32 * 16];
  __shared__ ScratchC S[8];
  const int tid = threadIdx.x;
  const int wave = tid >> 5, lane = tid & 31;
  const int m = lane & 15, kh = lane >> 4;

  // stage Wn_c (64x48) as packed bf16 pairs, 3 column tiles of 16
  for (int i = tid; i < 3 * 32 * 16; i += 256) {
    int n = i & 15, kp = (i >> 4) & 31, t = i >> 9;
    int col = t * 16 + n;
    unsigned lo = f2bf(Wn[(2 * kp) * 48 + col]);
    unsigned hi = f2bf(Wn[(2 * kp + 1) * 48 + col]);
    Bp[i] = lo | (hi << 16);
  }
  __syncthreads();

  ScratchC& W = S[wave];
  const int R = (blockIdx.x * 8 + wave) * 16;

  {  // gather: row m, channels strided by 32^3
    int r = R + m;
    int x = vi[4 * r + 0], y = vi[4 * r + 1], z = vi[4 * r + 2];
    int base = x * 1024 + y * 32 + z;
    for (int c = kh; c < 64; c += 2) W.F[m][c] = gc[c * 32768 + base];
  }
  __syncthreads();

  if (lane < 16) {  // layernorm -> bf16 A panel
    float s = 0.f, ss = 0.f;
#pragma unroll
    for (int k = 0; k < 64; ++k) { float v = W.F[lane][k]; s += v; ss += v * v; }
    float mean = s * (1.f / 64.f);
    float var  = ss * (1.f / 64.f) - mean * mean;
    float rs   = rsqrtf(var + 1e-5f);
#pragma unroll
    for (int k = 0; k < 64; ++k)
      W.A[lane][k] = f2bf((W.F[lane][k] - mean) * rs * lnw[k] + lnb[k]);
  }
  __syncthreads();

  v16bf a0 = load_a_frag(&W.A[0][0], 64, m, kh, 0);
  v16bf a1 = load_a_frag(&W.A[0][0], 64, m, kh, 32);
#pragma unroll
  for (int t = 0; t < 3; ++t) {
    v8f acc = {};
    acc = WMMA_BF16(a0, load_b_frag(Bp + t * 512, kh, m, 0), acc);
    acc = WMMA_BF16(a1, load_b_frag(Bp + t * 512, kh, m, 16), acc);
    float bias = bn[t * 16 + m];
#pragma unroll
    for (int v = 0; v < 8; ++v) {  // C/D layout: lane = col, VGPR v -> row 8*kh+v
      int row = 8 * kh + v;
      float val = acc[v] + bias;
      W.D[row][t * 16 + m] = val;
      vfc[(long)(R + row) * 48 + t * 16 + m] = val;
    }
  }
  __syncthreads();

  if (lane < 16) {  // occupancy logit
    float acc = bo[0];
#pragma unroll
    for (int c = 0; c < 48; ++c) acc += W.D[lane][c] * Wo[c];
    out_logc[R + lane] = acc;
  }
}

// ---------------------------------------------------------------------------
// Medium level: N1=262144 rows. concat[vf_c(48), LN(gather 32), log_c(1)] (K=81->96)
// @ Wn_m (96x24 padded, 2 col tiles) -> logit(24->1), valid_m
// ---------------------------------------------------------------------------
struct __align__(16) ScratchM {
  union { float F[16][33]; float D[16][24]; };
  unsigned short A[16][96];
};

__global__ __launch_bounds__(256) void spg_medium(
    const float* __restrict__ gm, const int* __restrict__ vi,
    const float* __restrict__ lnw, const float* __restrict__ lnb,
    const float* __restrict__ Wn, const float* __restrict__ bn,
    const float* __restrict__ Wo, const float* __restrict__ bo,
    const float* __restrict__ vfc, const float* __restrict__ logc,
    float* __restrict__ vfm, float* __restrict__ out_logm,
    float* __restrict__ out_vm) {
  __shared__ unsigned Bp[2 * 48 * 16];
  __shared__ ScratchM S[8];
  const int tid = threadIdx.x;
  const int wave = tid >> 5, lane = tid & 31;
  const int m = lane & 15, kh = lane >> 4;

  for (int i = tid; i < 2 * 48 * 16; i += 256) {
    int n = i & 15, kp = (i >> 4) % 48, t = i / (48 * 16);
    int col = t * 16 + n;
    int k0 = 2 * kp, k1 = k0 + 1;
    unsigned lo = (col < 24 && k0 < 81) ? (unsigned)f2bf(Wn[k0 * 24 + col]) : 0u;
    unsigned hi = (col < 24 && k1 < 81) ? (unsigned)f2bf(Wn[k1 * 24 + col]) : 0u;
    Bp[t * 768 + kp * 16 + n] = lo | (hi << 16);
  }
  __syncthreads();

  ScratchM& W = S[wave];
  const int R = (blockIdx.x * 8 + wave) * 16;
  const int r1 = R + m;
  const int r0 = r1 >> 3, j1 = r1 & 7;
  {
    int x0 = vi[4 * r0 + 0], y0 = vi[4 * r0 + 1], z0 = vi[4 * r0 + 2];
    int x1 = 2 * x0 + ((DXM >> j1) & 1);
    int y1 = 2 * y0 + ((DYM >> j1) & 1);
    int z1 = 2 * z0 + ((DZM >> j1) & 1);
    int base = x1 * 4096 + y1 * 64 + z1;
    for (int c = kh; c < 32; c += 2) W.F[m][c] = gm[c * 262144 + base];
    const float* src = vfc + (long)r0 * 48;      // upsampled coarse features
    for (int c = kh; c < 48; c += 2) W.A[m][c] = f2bf(src[c]);
  }
  __syncthreads();

  if (lane < 16) {
    float s = 0.f, ss = 0.f;
#pragma unroll
    for (int k = 0; k < 32; ++k) { float v = W.F[lane][k]; s += v; ss += v * v; }
    float mean = s * (1.f / 32.f);
    float var  = ss * (1.f / 32.f) - mean * mean;
    float rs   = rsqrtf(var + 1e-5f);
#pragma unroll
    for (int k = 0; k < 32; ++k)
      W.A[lane][48 + k] = f2bf((W.F[lane][k] - mean) * rs * lnw[k] + lnb[k]);
    W.A[lane][80] = f2bf(logc[(R + lane) >> 3]);  // log_prev
#pragma unroll
    for (int k = 81; k < 96; ++k) W.A[lane][k] = 0;  // K padding
  }
  __syncthreads();

  v16bf a0 = load_a_frag(&W.A[0][0], 96, m, kh, 0);
  v16bf a1 = load_a_frag(&W.A[0][0], 96, m, kh, 32);
  v16bf a2 = load_a_frag(&W.A[0][0], 96, m, kh, 64);
#pragma unroll
  for (int t = 0; t < 2; ++t) {
    v8f acc = {};
    acc = WMMA_BF16(a0, load_b_frag(Bp + t * 768, kh, m, 0), acc);
    acc = WMMA_BF16(a1, load_b_frag(Bp + t * 768, kh, m, 16), acc);
    acc = WMMA_BF16(a2, load_b_frag(Bp + t * 768, kh, m, 32), acc);
    int col = t * 16 + m;
    if (col < 24) {
      float bias = bn[col];
#pragma unroll
      for (int v = 0; v < 8; ++v) {
        int row = 8 * kh + v;
        float val = acc[v] + bias;
        W.D[row][col] = val;
        vfm[(long)(R + row) * 24 + col] = val;
      }
    }
  }
  __syncthreads();

  if (lane < 16) {
    int r = R + lane;
    float acc = bo[0];
#pragma unroll
    for (int c = 0; c < 24; ++c) acc += W.D[lane][c] * Wo[c];
    out_logm[r] = acc;
    out_vm[r] = (logc[r >> 3] > 0.f) ? 1.f : 0.f;
  }
}

// ---------------------------------------------------------------------------
// Dense level: N2=2097152 rows. concat[vf_m(24), LN(gather 16), log_m(1)] (K=41->64)
// @ Wn_d (64x16 padded) -> 10 task heads + valid_d
// ---------------------------------------------------------------------------
struct __align__(16) ScratchD {
  union { float F[16][17]; float D[16][16]; };
  unsigned short A[16][64];
};

__global__ __launch_bounds__(256) void spg_dense(
    const float* __restrict__ gd, const int* __restrict__ vi,
    const float* __restrict__ lnw, const float* __restrict__ lnb,
    const float* __restrict__ Wn, const float* __restrict__ bn,
    const float* __restrict__ Wo, const float* __restrict__ bo,
    const float* __restrict__ Woff, const float* __restrict__ boff,
    const float* __restrict__ Wrs, const float* __restrict__ brs,
    const float* __restrict__ Wrot, const float* __restrict__ brot,
    const float* __restrict__ Wwid, const float* __restrict__ bwid,
    const float* __restrict__ vfm, const float* __restrict__ logc,
    const float* __restrict__ logm,
    float* __restrict__ out_logd, float* __restrict__ out_off,
    float* __restrict__ out_rs, float* __restrict__ out_rot,
    float* __restrict__ out_wid, float* __restrict__ out_vd) {
  __shared__ unsigned Bp[32 * 16];
  __shared__ float Hw[16][10];
  __shared__ float Hb[10];
  __shared__ ScratchD S[8];
  const int tid = threadIdx.x;
  const int wave = tid >> 5, lane = tid & 31;
  const int m = lane & 15, kh = lane >> 4;

  for (int i = tid; i < 32 * 16; i += 256) {  // Wn_d padded 41->64 rows
    int n = i & 15, kp = i >> 4;
    int k0 = 2 * kp, k1 = k0 + 1;
    unsigned lo = (k0 < 41) ? (unsigned)f2bf(Wn[k0 * 16 + n]) : 0u;
    unsigned hi = (k1 < 41) ? (unsigned)f2bf(Wn[k1 * 16 + n]) : 0u;
    Bp[i] = lo | (hi << 16);
  }
  if (tid < 160) {  // fuse the 5 task heads into one 16x10 matrix
    int c = tid / 10, h = tid % 10;
    float w;
    if (h == 0)      w = Wo[c];
    else if (h < 4)  w = Woff[c * 3 + (h - 1)];
    else if (h == 4) w = Wrs[c];
    else if (h < 9)  w = Wrot[c * 4 + (h - 5)];
    else             w = Wwid[c];
    Hw[c][h] = w;
  }
  if (tid < 10) {
    float b;
    if (tid == 0)      b = bo[0];
    else if (tid < 4)  b = boff[tid - 1];
    else if (tid == 4) b = brs[0];
    else if (tid < 9)  b = brot[tid - 5];
    else               b = bwid[0];
    Hb[tid] = b;
  }
  __syncthreads();

  ScratchD& W = S[wave];
  const int R = (blockIdx.x * 8 + wave) * 16;
  const int r2 = R + m;
  const int r1 = r2 >> 3, j2 = r2 & 7;
  const int r0 = r1 >> 3, j1 = r1 & 7;
  {
    int x0 = vi[4 * r0 + 0], y0 = vi[4 * r0 + 1], z0 = vi[4 * r0 + 2];
    int x2 = 2 * (2 * x0 + ((DXM >> j1) & 1)) + ((DXM >> j2) & 1);
    int y2 = 2 * (2 * y0 + ((DYM >> j1) & 1)) + ((DYM >> j2) & 1);
    int z2 = 2 * (2 * z0 + ((DZM >> j1) & 1)) + ((DZM >> j2) & 1);
    long base = (long)x2 * 16384 + y2 * 128 + z2;
    for (int c = kh; c < 16; c += 2) W.F[m][c] = gd[(long)c * 2097152 + base];
    const float* src = vfm + (long)r1 * 24;      // upsampled medium features
    for (int c = kh; c < 24; c += 2) W.A[m][c] = f2bf(src[c]);
  }
  __syncthreads();

  if (lane < 16) {
    float s = 0.f, ss = 0.f;
#pragma unroll
    for (int k = 0; k < 16; ++k) { float v = W.F[lane][k]; s += v; ss += v * v; }
    float mean = s * (1.f / 16.f);
    float var  = ss * (1.f / 16.f) - mean * mean;
    float rs   = rsqrtf(var + 1e-5f);
#pragma unroll
    for (int k = 0; k < 16; ++k)
      W.A[lane][24 + k] = f2bf((W.F[lane][k] - mean) * rs * lnw[k] + lnb[k]);
    W.A[lane][40] = f2bf(logm[(R + lane) >> 3]);
#pragma unroll
    for (int k = 41; k < 64; ++k) W.A[lane][k] = 0;
  }
  __syncthreads();

  v16bf a0 = load_a_frag(&W.A[0][0], 64, m, kh, 0);
  v16bf a1 = load_a_frag(&W.A[0][0], 64, m, kh, 32);
  v8f acc = {};
  acc = WMMA_BF16(a0, load_b_frag(Bp, kh, m, 0), acc);
  acc = WMMA_BF16(a1, load_b_frag(Bp, kh, m, 16), acc);
  {
    float bias = bn[m];
#pragma unroll
    for (int v = 0; v < 8; ++v) W.D[8 * kh + v][m] = acc[v] + bias;
  }
  __syncthreads();

  if (lane < 16) {
    int r = R + lane;
    float vals[10];
#pragma unroll
    for (int h = 0; h < 10; ++h) vals[h] = Hb[h];
#pragma unroll
    for (int c = 0; c < 16; ++c) {
      float f = W.D[lane][c];
#pragma unroll
      for (int h = 0; h < 10; ++h) vals[h] += f * Hw[c][h];
    }
    out_logd[r] = vals[0];
    out_off[(long)r * 3 + 0] = vals[1];
    out_off[(long)r * 3 + 1] = vals[2];
    out_off[(long)r * 3 + 2] = vals[3];
    out_rs[r] = vals[4];
    out_rot[(long)r * 4 + 0] = vals[5];
    out_rot[(long)r * 4 + 1] = vals[6];
    out_rot[(long)r * 4 + 2] = vals[7];
    out_rot[(long)r * 4 + 3] = vals[8];
    out_wid[r] = vals[9];
    out_vd[r] = (logc[r >> 6] > 0.f && logm[r >> 3] > 0.f) ? 1.f : 0.f;
  }
}

// ---------------------------------------------------------------------------
extern "C" void kernel_launch(void* const* d_in, const int* in_sizes, int n_in,
                              void* d_out, int out_size, void* d_ws, size_t ws_size,
                              hipStream_t stream) {
  (void)in_sizes; (void)n_in; (void)out_size; (void)ws_size;
  const float* gc   = (const float*)d_in[0];
  const float* gm   = (const float*)d_in[1];
  const float* gd   = (const float*)d_in[2];
  const int*   vi   = (const int*)d_in[3];
  const float* lnwc = (const float*)d_in[4],  *lnbc = (const float*)d_in[5];
  const float* lnwm = (const float*)d_in[6],  *lnbm = (const float*)d_in[7];
  const float* lnwd = (const float*)d_in[8],  *lnbd = (const float*)d_in[9];
  const float* Wnc  = (const float*)d_in[10], *bnc  = (const float*)d_in[11];
  const float* Wnm  = (const float*)d_in[12], *bnm  = (const float*)d_in[13];
  const float* Wnd  = (const float*)d_in[14], *bnd  = (const float*)d_in[15];
  const float* Woc  = (const float*)d_in[16], *boc  = (const float*)d_in[17];
  const float* Wom  = (const float*)d_in[18], *bom  = (const float*)d_in[19];
  const float* Wod  = (const float*)d_in[20], *bod  = (const float*)d_in[21];
  const float* Woff = (const float*)d_in[22], *boff = (const float*)d_in[23];
  const float* Wrs  = (const float*)d_in[24], *brs  = (const float*)d_in[25];
  const float* Wrot = (const float*)d_in[26], *brot = (const float*)d_in[27];
  const float* Wwid = (const float*)d_in[28], *bwid = (const float*)d_in[29];

  const long N0 = 32768, N1 = N0 * 8, N2 = N1 * 8;
  float* out    = (float*)d_out;
  float* o_logc = out;
  float* o_logm = out + N0;
  float* o_logd = o_logm + N1;
  float* o_off  = o_logd + N2;
  float* o_rs   = o_off + 3 * N2;
  float* o_rot  = o_rs + N2;
  float* o_wid  = o_rot + 4 * N2;
  float* o_vm   = o_wid + N2;
  float* o_vd   = o_vm + N1;

  float* vfc = (float*)d_ws;          // [N0, 48]
  float* vfm = vfc + N0 * 48;         // [N1, 24]

  spg_coarse<<<dim3(256), dim3(256), 0, stream>>>(
      gc, vi, lnwc, lnbc, Wnc, bnc, Woc, boc, vfc, o_logc);
  spg_medium<<<dim3(2048), dim3(256), 0, stream>>>(
      gm, vi, lnwm, lnbm, Wnm, bnm, Wom, bom, vfc, o_logc, vfm, o_logm, o_vm);
  spg_dense<<<dim3(16384), dim3(256), 0, stream>>>(
      gd, vi, lnwd, lnbd, Wnd, bnd, Wod, bod, Woff, boff, Wrs, brs,
      Wrot, brot, Wwid, bwid, vfm, o_logc, o_logm,
      o_logd, o_off, o_rs, o_rot, o_wid, o_vd);
}
